// WindowAttention_16716012716556
// MI455X (gfx1250) — matrix operand order repaired and checked by reading, not compiled, not run on hardware
//
#include <hip/hip_runtime.h>
#include <hip/hip_bf16.h>

typedef __bf16 bf16_t;
typedef __attribute__((ext_vector_type(16))) __bf16 v16bf;
typedef __attribute__((ext_vector_type(8)))  float  v8f;

__device__ __forceinline__ v8f wmma_bf16(v16bf a, v16bf b, v8f c) {
  // (neg_a, A, neg_b, B, c_mod, C, reuse_a, reuse_b)
  return __builtin_amdgcn_wmma_f32_16x16x32_bf16(false, a, false, b, (short)0, c, false, false);
}

#define DIM   384
#define DIM3  1152        // 3*DIM, qkv row stride
#define WIN   7
#define HEADS 12
#define NTOK  49          // WIN*WIN
#define BW    4096
#define NW    64
#define HD    32          // head dim
#define MROWS (BW * NTOK) // 200704, divisible by 64

#define QKVW_E (3 * DIM * DIM)  // 442368
#define PRJW_E (DIM * DIM)      // 147456

// ---------------------------------------------------------------------------
// Kernel 0: one-shot weight conversion f32 -> bf16 (1.2 MB total, trivial).
// ---------------------------------------------------------------------------
__global__ __launch_bounds__(256)
void wprep(const float* __restrict__ qkv_w, const float* __restrict__ proj_w,
           bf16_t* __restrict__ wq, bf16_t* __restrict__ wp)
{
  const int i = blockIdx.x * 256 + threadIdx.x;
  if (i < QKVW_E) wq[i] = (bf16_t)qkv_w[i];
  if (i < PRJW_E) wp[i] = (bf16_t)proj_w[i];
}

// ---------------------------------------------------------------------------
// Kernel 1: QKV projection. qkv(row, col) = x @ qkv_w^T + qkv_b, bf16,
// stored in natural GEMM layout (row = b*49+n, col = which*384 + h*32 + d):
// no divisions / scatter in the epilogue, coalesced stores.
// q columns (col < 384) pre-scaled by 1/sqrt(d).
// Grid: (col-groups=18 FAST, row-groups=3136 SLOW) -> x row slabs reused in L2.
// ---------------------------------------------------------------------------
__global__ __launch_bounds__(128)
void qkv_gemm(const float* __restrict__ x, const bf16_t* __restrict__ w,
              const float* __restrict__ bias, bf16_t* __restrict__ qkv)
{
  const int lane = threadIdx.x & 31;
  const int wave = threadIdx.x >> 5;
  const int half = lane >> 4;
  const int l    = lane & 15;
  const int c0   = blockIdx.x * 64;                // col tile base (fast dim)
  const int m0   = (blockIdx.y * 4 + wave) * 16;   // row tile (slow dim)

  v8f acc[4] = {};
  for (int kk = 0; kk < DIM; kk += 32) {
    // Preload all four B fragments (pure bf16 32B loads), then A, then 4 WMMAs
    v16bf bfr[4];
#pragma unroll
    for (int j = 0; j < 4; ++j) {
      const bf16_t* wr = w + (size_t)(c0 + 16 * j + l) * DIM + kk + half * 16;
#pragma unroll
      for (int t = 0; t < 16; ++t) bfr[j][t] = wr[t];
    }
    // A fragment (16x32 bf16): lane<16 holds row m0+l, K {0..15}
    v16bf a;
    {
      const float* xr = x + (size_t)(m0 + l) * DIM + kk;
      const int klo = half * 16, khi = half * 16 + 8;
#pragma unroll
      for (int t = 0; t < 8; ++t) a[t]     = (bf16_t)xr[klo + t];
#pragma unroll
      for (int t = 0; t < 8; ++t) a[8 + t] = (bf16_t)xr[khi + t];
    }
#pragma unroll
    for (int j = 0; j < 4; ++j)
      acc[j] = wmma_bf16(a, bfr[j], acc[j]);
  }

  const float scale = 0.17677669529663687f;  // 1/sqrt(32)
#pragma unroll
  for (int j = 0; j < 4; ++j) {
    const int col = c0 + 16 * j + l;
    const float bv = bias[col];
    const float sc = (col < DIM) ? scale : 1.0f;   // q columns pre-scaled
#pragma unroll
    for (int v = 0; v < 8; ++v) {
      const int row = m0 + v + half * 8;           // D layout
      qkv[(size_t)row * DIM3 + col] = (bf16_t)((acc[j][v] + bv) * sc);
    }
  }
}

// ---------------------------------------------------------------------------
// Kernel 2: fused attention. One wave per (window b, head h).
// S = qk^T (4x4 WMMA tiles over 64x64 padded), +bias+mask, row softmax via
// shfl_xor (width-16 halves == D-layout rows), P->LDS (bf16, stride 80),
// O = P @ V (2 K-steps x 2 col tiles), O -> ws bf16 (b*n, 384).
// Q/K/V read from the (row, 1152) qkv buffer with row stride 1152.
// ---------------------------------------------------------------------------
#define LDSTR 80   // bf16 elements per LDS row: keeps 8-elt reads 16B aligned

__global__ __launch_bounds__(128)
void attn_kernel(const bf16_t* __restrict__ qkv,
                 const float* __restrict__ bias_table,
                 const float* __restrict__ mask,
                 const int* __restrict__ rel_index,
                 bf16_t* __restrict__ ows)
{
  __shared__ bf16_t Plds[4 * 64 * LDSTR];   // 40 KB, one 64x80 panel per wave
  const int lane = threadIdx.x & 31;
  const int wave = threadIdx.x >> 5;
  const int half = lane >> 4;
  const int l    = lane & 15;
  const int b    = blockIdx.x;
  const int h    = blockIdx.y * 4 + wave;

  bf16_t* P = Plds + wave * 64 * LDSTR;
  const bf16_t* base = qkv + (size_t)b * NTOK * DIM3 + h * HD;
  const bf16_t* Q = base;              // cols [h*32, h*32+32)
  const bf16_t* K = base + DIM;        // + 384
  const bf16_t* V = base + 2 * DIM;    // + 768
  const float*  Mk = mask + (size_t)(b & (NW - 1)) * NTOK * NTOK;

  // K fragments: B-matrix 32x16 per tile j; "col" = key row index
  v16bf kb[4];
#pragma unroll
  for (int j = 0; j < 4; ++j) {
    const int r = 16 * j + l;
    if (r < NTOK) {
      const bf16_t* p = K + (size_t)r * DIM3 + half * 16;
#pragma unroll
      for (int t = 0; t < 16; ++t) kb[j][t] = p[t];
    } else {
#pragma unroll
      for (int t = 0; t < 16; ++t) kb[j][t] = (bf16_t)0.0f;
    }
  }

  for (int i = 0; i < 4; ++i) {
    // Q A-fragment for rows 16i..16i+15
    v16bf qa;
    {
      const int r = 16 * i + l;
      if (r < NTOK) {
        const bf16_t* p = Q + (size_t)r * DIM3;
#pragma unroll
        for (int t = 0; t < 8; ++t) qa[t]     = p[half * 16 + t];
#pragma unroll
        for (int t = 0; t < 8; ++t) qa[8 + t] = p[half * 16 + 8 + t];
      } else {
#pragma unroll
        for (int t = 0; t < 16; ++t) qa[t] = (bf16_t)0.0f;
      }
    }

    v8f s[4];
#pragma unroll
    for (int j = 0; j < 4; ++j) {
      v8f z = {};
      s[j] = wmma_bf16(qa, kb[j], z);
    }

    // bias + mask + padding mask, in D layout
#pragma unroll
    for (int j = 0; j < 4; ++j) {
#pragma unroll
      for (int v = 0; v < 8; ++v) {
        const int row = 16 * i + v + half * 8;
        const int col = 16 * j + l;
        float val;
        if (row < NTOK && col < NTOK) {
          const int ridx = rel_index[row * NTOK + col];
          val = s[j][v] + bias_table[ridx * HEADS + h] + Mk[row * NTOK + col];
        } else {
          val = -1e30f;
        }
        s[j][v] = val;
      }
    }

    // row softmax: reductions across lanes 0..15 / 16..31 (xor 1..8 stays in half)
#pragma unroll
    for (int v = 0; v < 8; ++v) {
      float m = fmaxf(fmaxf(s[0][v], s[1][v]), fmaxf(s[2][v], s[3][v]));
#pragma unroll
      for (int off = 8; off >= 1; off >>= 1) m = fmaxf(m, __shfl_xor(m, off, 32));
      float e[4];
      float sum = 0.0f;
#pragma unroll
      for (int j = 0; j < 4; ++j) { e[j] = __expf(s[j][v] - m); sum += e[j]; }
#pragma unroll
      for (int off = 8; off >= 1; off >>= 1) sum += __shfl_xor(sum, off, 32);
      const float rinv = 1.0f / sum;
      const int row = 16 * i + v + half * 8;
#pragma unroll
      for (int j = 0; j < 4; ++j) {
        const float pv = (row < NTOK) ? e[j] * rinv : 0.0f;
        P[row * LDSTR + 16 * j + l] = (bf16_t)pv;
      }
    }
  }

  // V B-fragments: K-dim = key index (2 chunks of 32), N = head dim (2 tiles)
  v16bf vb[2][2];
#pragma unroll
  for (int kk2 = 0; kk2 < 2; ++kk2) {
#pragma unroll
    for (int j = 0; j < 2; ++j) {
#pragma unroll
      for (int t = 0; t < 16; ++t) {
        const int krow = kk2 * 32 + half * 16 + t;
        const int dcol = 16 * j + l;
        vb[kk2][j][t] = (krow < NTOK) ? V[(size_t)krow * DIM3 + dcol] : (bf16_t)0.0f;
      }
    }
  }

  for (int i = 0; i < 4; ++i) {
    // P A-fragments from LDS (contiguous 16B reads)
    v16bf pa[2];
#pragma unroll
    for (int kk2 = 0; kk2 < 2; ++kk2) {
      const bf16_t* p = P + (16 * i + l) * LDSTR + kk2 * 32;
#pragma unroll
      for (int t = 0; t < 8; ++t) pa[kk2][t]     = p[half * 16 + t];
#pragma unroll
      for (int t = 0; t < 8; ++t) pa[kk2][8 + t] = p[half * 16 + 8 + t];
    }
#pragma unroll
    for (int j = 0; j < 2; ++j) {
      v8f o = {};
      o = wmma_bf16(pa[0], vb[0][j], o);
      o = wmma_bf16(pa[1], vb[1][j], o);
#pragma unroll
      for (int v = 0; v < 8; ++v) {
        const int row = 16 * i + v + half * 8;
        if (row < NTOK)
          ows[((size_t)b * NTOK + row) * DIM + h * HD + 16 * j + l] = (bf16_t)o[v];
      }
    }
  }
}

// ---------------------------------------------------------------------------
// Kernel 3: output projection. out(f32) = O(bf16) @ proj_w^T + proj_b.
// Grid: (col-groups=6 FAST, row-groups SLOW) for L2 reuse of O rows.
// ---------------------------------------------------------------------------
__global__ __launch_bounds__(128)
void proj_gemm(const bf16_t* __restrict__ ows, const bf16_t* __restrict__ w,
               const float* __restrict__ bias, float* __restrict__ out)
{
  const int lane = threadIdx.x & 31;
  const int wave = threadIdx.x >> 5;
  const int half = lane >> 4;
  const int l    = lane & 15;
  const int c0   = blockIdx.x * 64;                // fast dim
  const int m0   = (blockIdx.y * 4 + wave) * 16;   // slow dim

  v8f acc[4] = {};
  for (int kk = 0; kk < DIM; kk += 32) {
    v16bf bfr[4];
#pragma unroll
    for (int j = 0; j < 4; ++j) {
      const bf16_t* wr = w + (size_t)(c0 + 16 * j + l) * DIM + kk + half * 16;
#pragma unroll
      for (int t = 0; t < 16; ++t) bfr[j][t] = wr[t];
    }
    v16bf a;
    {
      const bf16_t* xr = ows + (size_t)(m0 + l) * DIM + kk;
#pragma unroll
      for (int t = 0; t < 8; ++t) a[t]     = xr[half * 16 + t];
#pragma unroll
      for (int t = 0; t < 8; ++t) a[8 + t] = xr[half * 16 + 8 + t];
    }
#pragma unroll
    for (int j = 0; j < 4; ++j)
      acc[j] = wmma_bf16(a, bfr[j], acc[j]);
  }
#pragma unroll
  for (int j = 0; j < 4; ++j) {
    const int col = c0 + 16 * j + l;
    const float bv = bias[col];
#pragma unroll
    for (int v = 0; v < 8; ++v) {
      const int row = m0 + v + half * 8;
      out[(size_t)row * DIM + col] = acc[j][v] + bv;
    }
  }
}

// ---------------------------------------------------------------------------
extern "C" void kernel_launch(void* const* d_in, const int* in_sizes, int n_in,
                              void* d_out, int out_size, void* d_ws, size_t ws_size,
                              hipStream_t stream) {
  const float* x          = (const float*)d_in[0];
  const float* mask       = (const float*)d_in[1];
  const float* bias_table = (const float*)d_in[2];
  const float* qkv_w      = (const float*)d_in[3];
  const float* qkv_b      = (const float*)d_in[4];
  const float* proj_w     = (const float*)d_in[5];
  const float* proj_b     = (const float*)d_in[6];
  const int*   rel_index  = (const int*)d_in[7];
  float* out = (float*)d_out;

  const size_t QKV_E = (size_t)MROWS * DIM3;   // 231,211,008 bf16 elems
  const size_t O_E   = (size_t)MROWS * DIM;    //  77,070,336 bf16 elems
  bf16_t* qkvws = (bf16_t*)d_ws;               // (row, 1152)
  bf16_t* ows   = qkvws + QKV_E;               // (row, 384)
  bf16_t* wq    = ows + O_E;                   // bf16 qkv_w (1152x384)
  bf16_t* wp    = wq + QKVW_E;                 // bf16 proj_w (384x384)

  // 0) Convert weights to bf16 once.
  wprep<<<(QKVW_E + 255) / 256, 256, 0, stream>>>(qkv_w, proj_w, wq, wp);

  // 1) QKV projection. Col groups fastest (18), row groups slow (3136).
  qkv_gemm<<<dim3(DIM3 / 64, MROWS / 64), 128, 0, stream>>>(
      x, wq, qkv_b, qkvws);

  // 2) Attention: one wave per (window, head): 4096 x (12/4) blocks.
  attn_kernel<<<dim3(BW, HEADS / 4), 128, 0, stream>>>(
      qkvws, bias_table, mask, rel_index, ows);

  // 3) Output projection. Col groups fastest (6), row groups slow.
  proj_gemm<<<dim3(DIM / 64, MROWS / 64), 128, 0, stream>>>(
      ows, wp, proj_b, out);
}
